// Block_50225347559499
// MI455X (gfx1250) — matrix operand order repaired and checked by reading, not compiled, not run on hardware
//
#include <hip/hip_runtime.h>
#include <hip/hip_bf16.h>
#include <math.h>

// ---------------------------------------------------------------- types
typedef __bf16 bf16_t;
typedef __attribute__((ext_vector_type(16))) __bf16 v16bf;
typedef __attribute__((ext_vector_type(8)))  float  v8f;

union FragB16 {
  v16bf v;
  uint4 q[2];
  bf16_t e[16];
};

__device__ __forceinline__ v8f wmma_bf16(v16bf a, v16bf b, v8f c) {
  // (neg_a, A, neg_b, B, c_mod, C, reuse_a, reuse_b)
  return __builtin_amdgcn_wmma_f32_16x16x32_bf16(false, a, false, b, (short)0, c,
                                                 false, false);
}

// A-matrix fragment (16x32 bf16, M x K). Lane r (=lane&15) holds row r.
// lanes 0-15: K chunks [k0+0..7],[k0+16..23]; lanes 16-31: [k0+8..15],[k0+24..31]
__device__ __forceinline__ v16bf load_frag_a(const bf16_t* base, int ld, int k0) {
  const int lane = threadIdx.x & 31;
  const int r = lane & 15, sel = lane >> 4;
  const bf16_t* p = base + (size_t)r * ld + k0 + sel * 8;
  FragB16 f;
  f.q[0] = *(const uint4*)p;
  f.q[1] = *(const uint4*)(p + 16);
  return f.v;
}

// B-matrix fragment (32x16 bf16, K x N), sourced from row-major B^T (N x K).
// Lane c (=lane&15) holds column c; lanes 0-15: K chunks [k0+0..7],[k0+16..23]; lanes 16-31: [k0+8..15],[k0+24..31]
__device__ __forceinline__ v16bf load_frag_b(const bf16_t* baseT, int ld, int k0) {
  const int lane = threadIdx.x & 31;
  const int c = lane & 15, sel = lane >> 4;
  const bf16_t* p = baseT + (size_t)c * ld + k0 + sel * 8;
  FragB16 f;
  f.q[0] = *(const uint4*)p;
  f.q[1] = *(const uint4*)(p + 16);
  return f.v;
}

__device__ __forceinline__ float gelu_exact(float v) {
  return 0.5f * v * (1.0f + erff(v * 0.70710678118654752f));
}

// ---------------------------------------------------------------- weight prep
// W [K x N] fp32 row-major  ->  WT [N x K] bf16 row-major
__global__ __launch_bounds__(256) void transpose_bf16_kernel(
    const float* __restrict__ W, bf16_t* __restrict__ WT, int K, int N) {
  __shared__ float t[32][33];
  const int tx = threadIdx.x, ty = threadIdx.y;
  const int n0 = blockIdx.x * 32, k0 = blockIdx.y * 32;
#pragma unroll
  for (int j = 0; j < 4; ++j)
    t[ty + j * 8][tx] = W[(size_t)(k0 + ty + j * 8) * N + n0 + tx];
  __syncthreads();
#pragma unroll
  for (int j = 0; j < 4; ++j)
    WT[(size_t)(n0 + ty + j * 8) * K + k0 + tx] = (bf16_t)t[tx][ty + j * 8];
}

// ---------------------------------------------------------------- layernorm
// one 256-thread block per row of 768; fp32 in -> bf16 out
__global__ __launch_bounds__(256) void ln_kernel(
    const float* __restrict__ x, const float* __restrict__ g,
    const float* __restrict__ bta, bf16_t* __restrict__ out) {
  __shared__ float ssum[256], ssq[256];
  const int tid = threadIdx.x;
  const size_t base = (size_t)blockIdx.x * 768;
  const float v0 = x[base + tid];
  const float v1 = x[base + tid + 256];
  const float v2 = x[base + tid + 512];
  ssum[tid] = v0 + v1 + v2;
  ssq[tid]  = v0 * v0 + v1 * v1 + v2 * v2;
  __syncthreads();
  for (int o = 128; o > 0; o >>= 1) {
    if (tid < o) { ssum[tid] += ssum[tid + o]; ssq[tid] += ssq[tid + o]; }
    __syncthreads();
  }
  const float mu = ssum[0] * (1.0f / 768.0f);
  const float var = ssq[0] * (1.0f / 768.0f) - mu * mu;
  const float rs = rsqrtf(var + 1e-5f);
  out[base + tid]       = (bf16_t)((v0 - mu) * rs * g[tid]       + bta[tid]);
  out[base + tid + 256] = (bf16_t)((v1 - mu) * rs * g[tid + 256] + bta[tid + 256]);
  out[base + tid + 512] = (bf16_t)((v2 - mu) * rs * g[tid + 512] + bta[tid + 512]);
}

// ---------------------------------------------------------------- WMMA GEMM
// C[M,N] = act(A[M,K](bf16) @ BT[N,K]^T(bf16) + bias) (+ resid)
// 128 threads = 4 waves (2x2); block tile 64x128; wave tile 32x64 (2x4 WMMA).
// OUT_MODE: 0 = bf16 row-major, 1 = fp32 + residual, 2 = bf16 scattered to
//           per-head transposed layout vt[b,h,d,n] (for attention P@V loads).
template <int ACT_GELU, int OUT_MODE>
__global__ __launch_bounds__(128) void gemm_bf16_kernel(
    const bf16_t* __restrict__ A, const bf16_t* __restrict__ BT,
    const float* __restrict__ bias, const float* __restrict__ resid,
    float* __restrict__ outF, bf16_t* __restrict__ outB, int M, int N, int K) {
  const int nb = N >> 7;                 // 128-wide block columns
  const int bm = blockIdx.x / nb;
  const int bn = blockIdx.x - bm * nb;
  const int w = threadIdx.x >> 5;
  const int lane = threadIdx.x & 31;
  const int lr = lane & 15, hi = lane >> 4;
  const int row0 = bm * 64 + (w >> 1) * 32;
  const int col0 = bn * 128 + (w & 1) * 64;

  v8f acc[2][4];
#pragma unroll
  for (int j = 0; j < 4; ++j) {
    const float bj = bias[col0 + j * 16 + lr];
#pragma unroll
    for (int r = 0; r < 8; ++r) { acc[0][j][r] = bj; acc[1][j][r] = bj; }
  }
  const bf16_t* a0 = A + (size_t)row0 * K;
  const bf16_t* a1 = A + (size_t)(row0 + 16) * K;
  const bf16_t* bp[4];
#pragma unroll
  for (int j = 0; j < 4; ++j) bp[j] = BT + (size_t)(col0 + j * 16) * K;

  for (int k0 = 0; k0 < K; k0 += 32) {
    v16bf fa0 = load_frag_a(a0, K, k0);
    v16bf fa1 = load_frag_a(a1, K, k0);
    v16bf fb0 = load_frag_b(bp[0], K, k0);
    v16bf fb1 = load_frag_b(bp[1], K, k0);
    v16bf fb2 = load_frag_b(bp[2], K, k0);
    v16bf fb3 = load_frag_b(bp[3], K, k0);
    // hint next K-tile into cache (global_prefetch_b8)
    __builtin_prefetch(a0 + (size_t)lr * K + k0 + 32, 0, 1);
    __builtin_prefetch(bp[hi * 2] + (size_t)lr * K + k0 + 32, 0, 1);
    acc[0][0] = wmma_bf16(fa0, fb0, acc[0][0]);
    acc[0][1] = wmma_bf16(fa0, fb1, acc[0][1]);
    acc[0][2] = wmma_bf16(fa0, fb2, acc[0][2]);
    acc[0][3] = wmma_bf16(fa0, fb3, acc[0][3]);
    acc[1][0] = wmma_bf16(fa1, fb0, acc[1][0]);
    acc[1][1] = wmma_bf16(fa1, fb1, acc[1][1]);
    acc[1][2] = wmma_bf16(fa1, fb2, acc[1][2]);
    acc[1][3] = wmma_bf16(fa1, fb3, acc[1][3]);
  }

#pragma unroll
  for (int i = 0; i < 2; ++i)
#pragma unroll
    for (int j = 0; j < 4; ++j)
#pragma unroll
      for (int r = 0; r < 8; ++r) {
        const int mr = row0 + i * 16 + r + 8 * hi;     // C-layout: M = r + 8*hi
        const int nc = col0 + j * 16 + lr;             //           N = lane&15
        float v = acc[i][j][r];
        if (OUT_MODE == 1) {
          const size_t idx = (size_t)mr * N + nc;
          outF[idx] = resid[idx] + v;
        } else if (OUT_MODE == 2) {
          // vt[((b*12 + h)*64 + d) * 1024 + n]
          const int bb = mr >> 10, n = mr & 1023;
          const int h = nc >> 6, dd = nc & 63;
          outB[(((size_t)bb * 12 + h) * 64 + dd) * 1024 + n] = (bf16_t)v;
        } else {
          if (ACT_GELU) v = gelu_exact(v);
          outB[(size_t)mr * N + nc] = (bf16_t)v;
        }
      }
}

// ---------------------------------------------------------------- attention
// Flash-style: one wave per 16-query tile; d=64; online softmax over 1024 keys.
// V is consumed from the d-major layout vt[b,h,d,n] so P@V B-fragments are
// contiguous b128 loads.
__global__ __launch_bounds__(128) void attn_kernel(
    const bf16_t* __restrict__ qb, const bf16_t* __restrict__ kb,
    const bf16_t* __restrict__ vt, bf16_t* __restrict__ ob) {
  constexpr int NN = 1024, CC = 768, HH = 12, DD = 64;
  __shared__ __align__(16) bf16_t pl[4][16 * 32];

  const int w = threadIdx.x >> 5;
  const int tile = blockIdx.x * 4 + w;               // 6144 tiles total
  const int b = tile / (HH * (NN / 16));
  const int rem = tile - b * (HH * (NN / 16));
  const int h = rem / (NN / 16);
  const int qt = rem - h * (NN / 16);
  const int lane = threadIdx.x & 31;
  const int lr = lane & 15, sel = lane >> 4;

  const bf16_t* qbase = qb + ((size_t)b * NN + qt * 16) * CC + h * DD;
  const bf16_t* kbase = kb + ((size_t)b * NN) * CC + h * DD;
  const bf16_t* vtbase = vt + (((size_t)b * HH + h) * DD) * NN;  // [d][n]

  // resident Q fragments (16 x 64 = two K-chunks)
  const v16bf aq0 = load_frag_a(qbase, CC, 0);
  const v16bf aq1 = load_frag_a(qbase, CC, 32);

  v8f accO[4];
  float m[8], l[8];
#pragma unroll
  for (int t = 0; t < 4; ++t)
#pragma unroll
    for (int r = 0; r < 8; ++r) accO[t][r] = 0.0f;
#pragma unroll
  for (int r = 0; r < 8; ++r) { m[r] = -1e30f; l[r] = 0.0f; }

  const float scale = 0.125f;  // 1/sqrt(64)

  for (int kt = 0; kt < NN / 32; ++kt) {
    const int kr = kt * 32;
    // S = Q @ K^T for this 16x32 key tile (two 16x16 sub-tiles)
    v16bf bk00 = load_frag_b(kbase + (size_t)kr * CC, CC, 0);
    v16bf bk01 = load_frag_b(kbase + (size_t)kr * CC, CC, 32);
    v16bf bk10 = load_frag_b(kbase + (size_t)(kr + 16) * CC, CC, 0);
    v16bf bk11 = load_frag_b(kbase + (size_t)(kr + 16) * CC, CC, 32);
    __builtin_prefetch(kbase + (size_t)(kr + 32 + lr) * CC, 0, 1);
    __builtin_prefetch(vtbase + (size_t)(sel * 32 + lr) * NN + kr + 32, 0, 1);
    v8f s0, s1;
#pragma unroll
    for (int r = 0; r < 8; ++r) { s0[r] = 0.0f; s1[r] = 0.0f; }
    s0 = wmma_bf16(aq0, bk00, s0);
    s0 = wmma_bf16(aq1, bk01, s0);
    s1 = wmma_bf16(aq0, bk10, s1);
    s1 = wmma_bf16(aq1, bk11, s1);

    // online softmax; row M = r + 8*sel lives in the lane-half, reduce over 16 lanes
    float alpha[8];
#pragma unroll
    for (int r = 0; r < 8; ++r) {
      const float x0 = s0[r] * scale;
      const float x1 = s1[r] * scale;
      float mx = fmaxf(x0, x1);
      mx = fmaxf(mx, __shfl_xor(mx, 1, 32));
      mx = fmaxf(mx, __shfl_xor(mx, 2, 32));
      mx = fmaxf(mx, __shfl_xor(mx, 4, 32));
      mx = fmaxf(mx, __shfl_xor(mx, 8, 32));
      const float mn = fmaxf(m[r], mx);
      alpha[r] = __expf(m[r] - mn);
      m[r] = mn;
      const float p0 = __expf(x0 - mn);
      const float p1 = __expf(x1 - mn);
      float rs = p0 + p1;
      rs += __shfl_xor(rs, 1, 32);
      rs += __shfl_xor(rs, 2, 32);
      rs += __shfl_xor(rs, 4, 32);
      rs += __shfl_xor(rs, 8, 32);
      l[r] = l[r] * alpha[r] + rs;
      // stash P (bf16) into wave-private LDS slab, C-layout -> row-major 16x32
      pl[w][(r + 8 * sel) * 32 + lr]      = (bf16_t)p0;
      pl[w][(r + 8 * sel) * 32 + 16 + lr] = (bf16_t)p1;
    }
#pragma unroll
    for (int t = 0; t < 4; ++t)
#pragma unroll
      for (int r = 0; r < 8; ++r) accO[t][r] *= alpha[r];

    // reload P as an A fragment (same-wave DS ops are in-order)
    FragB16 fp;
    const bf16_t* pp = &pl[w][lr * 32 + sel * 8];
    fp.q[0] = *(const uint4*)pp;
    fp.q[1] = *(const uint4*)(pp + 16);

    // O += P @ V  (four d-tiles of 16); Vt rows are keys-contiguous -> b128 loads
#pragma unroll
    for (int j = 0; j < 4; ++j) {
      v16bf fv = load_frag_b(vtbase + (size_t)j * 16 * NN, NN, kr);
      accO[j] = wmma_bf16(fp.v, fv, accO[j]);
    }
  }

  // normalize and store (C-layout)
#pragma unroll
  for (int t = 0; t < 4; ++t)
#pragma unroll
    for (int r = 0; r < 8; ++r) {
      const float o = accO[t][r] / l[r];
      const size_t row = (size_t)b * NN + qt * 16 + r + 8 * sel;
      ob[row * CC + h * DD + t * 16 + lr] = (bf16_t)o;
    }
}

// ---------------------------------------------------------------- launch
extern "C" void kernel_launch(void* const* d_in, const int* in_sizes, int n_in,
                              void* d_out, int out_size, void* d_ws, size_t ws_size,
                              hipStream_t stream) {
  (void)in_sizes; (void)n_in; (void)out_size; (void)ws_size;
  const float* x    = (const float*)d_in[0];
  const float* ln1g = (const float*)d_in[1];
  const float* ln1b = (const float*)d_in[2];
  const float* Wq   = (const float*)d_in[3];
  const float* bq   = (const float*)d_in[4];
  const float* Wk   = (const float*)d_in[5];
  const float* bk   = (const float*)d_in[6];
  const float* Wv   = (const float*)d_in[7];
  const float* bv   = (const float*)d_in[8];
  const float* Wo   = (const float*)d_in[9];
  const float* bo   = (const float*)d_in[10];
  const float* ln2g = (const float*)d_in[11];
  const float* ln2b = (const float*)d_in[12];
  const float* W1   = (const float*)d_in[13];
  const float* b1   = (const float*)d_in[14];
  const float* W2   = (const float*)d_in[15];
  const float* b2   = (const float*)d_in[16];
  float* out = (float*)d_out;

  const int Bn = 8, Nn = 1024, C = 768, Hh = 3072;
  const int M = Bn * Nn;                       // 8192
  const size_t S  = (size_t)M * C;             // 6,291,456
  const size_t SH = (size_t)M * Hh;

  char* p = (char*)d_ws;
  auto alloc = [&](size_t bytes) -> char* {
    char* r = p;
    p += (bytes + 255) & ~(size_t)255;
    return r;
  };
  bf16_t* WqT  = (bf16_t*)alloc((size_t)C * C * 2);
  bf16_t* WkT  = (bf16_t*)alloc((size_t)C * C * 2);
  bf16_t* WvT  = (bf16_t*)alloc((size_t)C * C * 2);
  bf16_t* WoT  = (bf16_t*)alloc((size_t)C * C * 2);
  bf16_t* W1T  = (bf16_t*)alloc((size_t)C * Hh * 2);
  bf16_t* W2T  = (bf16_t*)alloc((size_t)C * Hh * 2);
  bf16_t* xnb  = (bf16_t*)alloc(S * 2);        // reused for ln1 and ln2 outputs
  bf16_t* qb   = (bf16_t*)alloc(S * 2);
  bf16_t* kb   = (bf16_t*)alloc(S * 2);
  bf16_t* vt   = (bf16_t*)alloc(S * 2);        // V in [b,h,d,n] layout
  bf16_t* obuf = (bf16_t*)alloc(S * 2);
  float*  x1   = (float*)alloc(S * 4);
  bf16_t* hb   = (bf16_t*)alloc(SH * 2);

  // weights -> bf16 transposed
  dim3 tb(32, 8);
  transpose_bf16_kernel<<<dim3(C / 32, C / 32), tb, 0, stream>>>(Wq, WqT, C, C);
  transpose_bf16_kernel<<<dim3(C / 32, C / 32), tb, 0, stream>>>(Wk, WkT, C, C);
  transpose_bf16_kernel<<<dim3(C / 32, C / 32), tb, 0, stream>>>(Wv, WvT, C, C);
  transpose_bf16_kernel<<<dim3(C / 32, C / 32), tb, 0, stream>>>(Wo, WoT, C, C);
  transpose_bf16_kernel<<<dim3(Hh / 32, C / 32), tb, 0, stream>>>(W1, W1T, C, Hh);
  transpose_bf16_kernel<<<dim3(C / 32, Hh / 32), tb, 0, stream>>>(W2, W2T, Hh, C);

  // attention branch
  ln_kernel<<<M, 256, 0, stream>>>(x, ln1g, ln1b, xnb);
  const int gCC = (M / 64) * (C / 128);
  gemm_bf16_kernel<0, 0><<<gCC, 128, 0, stream>>>(xnb, WqT, bq, nullptr, nullptr, qb, M, C, C);
  gemm_bf16_kernel<0, 0><<<gCC, 128, 0, stream>>>(xnb, WkT, bk, nullptr, nullptr, kb, M, C, C);
  gemm_bf16_kernel<0, 2><<<gCC, 128, 0, stream>>>(xnb, WvT, bv, nullptr, nullptr, vt, M, C, C);
  attn_kernel<<<(Bn * 12 * (Nn / 16)) / 4, 128, 0, stream>>>(qb, kb, vt, obuf);
  gemm_bf16_kernel<0, 1><<<gCC, 128, 0, stream>>>(obuf, WoT, bo, x, x1, nullptr, M, C, C);

  // MLP branch
  ln_kernel<<<M, 256, 0, stream>>>(x1, ln2g, ln2b, xnb);
  gemm_bf16_kernel<1, 0><<<(M / 64) * (Hh / 128), 128, 0, stream>>>(xnb, W1T, b1, nullptr, nullptr, hb, M, Hh, C);
  gemm_bf16_kernel<0, 1><<<gCC, 128, 0, stream>>>(hb, W2T, b2, x1, out, nullptr, M, C, Hh);
}
